// VarianceAdaptor_24300924961211
// MI455X (gfx1250) — compile-verified, matrix operand order checked by reading, not run on hardware
//
#include <hip/hip_runtime.h>
#include <hip/hip_bf16.h>

// ---------------------------------------------------------------------------
// VarianceAdaptor (FastSpeech2) for MI455X / gfx1250.
// Conv1d(k=3,256->256) passes run as bf16 WMMA GEMMs (v_wmma_f32_16x16x32_bf16)
// with fused bias + LayerNorm + ReLU (+ Linear head) epilogues.
// Activation tiles staged to LDS via GLOBAL_LOAD_ASYNC_TO_LDS when available.
// ---------------------------------------------------------------------------

#define B_  32
#define L_  1024
#define D_  256
#define T_  3072
#define NBINS_ 255   // NB - 1

typedef __bf16 v16bf __attribute__((ext_vector_type(16)));
typedef float  v8f   __attribute__((ext_vector_type(8)));

#if defined(__gfx1250__) && __has_builtin(__builtin_amdgcn_global_load_async_to_lds_b128)
#define HAVE_ASYNC_LDS 1
// Builtin signature (from compiler diagnostics): arg0 = int4 vector pointer in
// the global address space, arg1 = int4 vector pointer in LDS, then offset/cpol.
typedef int v4i_ __attribute__((vector_size(16)));
typedef __attribute__((address_space(1))) v4i_* gas_v4i;
typedef __attribute__((address_space(3))) v4i_* las_v4i;
#else
#define HAVE_ASYNC_LDS 0
#endif

__device__ __forceinline__ void wait_asynccnt0() {
#if __has_builtin(__builtin_amdgcn_s_wait_asynccnt)
    __builtin_amdgcn_s_wait_asynccnt(0);
#else
    asm volatile("s_wait_asynccnt 0" ::: "memory");
#endif
}

__device__ __forceinline__ unsigned short f2bf(float f) {
    unsigned int u = __float_as_uint(f);
    u += 0x7FFFu + ((u >> 16) & 1u);       // round-to-nearest-even
    return (unsigned short)(u >> 16);
}

// ---------------------------------------------------------------------------
// float -> bf16 cast (x tensor)
// ---------------------------------------------------------------------------
__global__ void cvt_bf16_kernel(const float* __restrict__ in,
                                unsigned short* __restrict__ out, int n) {
    int t = blockIdx.x * 256 + threadIdx.x;
    if (t < n) out[t] = f2bf(in[t]);
}

// ---------------------------------------------------------------------------
// Conv weights [K][Din][Dout] f32 -> transposed bf16 [K][Dout][Din]
// so B-matrix fragments are contiguous 32B per lane.
// ---------------------------------------------------------------------------
__global__ void cvt_wT_kernel(const float* __restrict__ w,
                              unsigned short* __restrict__ wT) {
    int t = blockIdx.x * 256 + threadIdx.x;
    if (t >= 3 * D_ * D_) return;
    int k   = t / (D_ * D_);
    int rem = t - k * (D_ * D_);
    int i   = rem >> 8;     // d_in
    int o   = rem & 255;    // d_out
    wT[((size_t)k * D_ + o) * D_ + i] = f2bf(w[t]);
}

// ---------------------------------------------------------------------------
// Per-batch inclusive scan of durations (Hillis-Steele, one block per batch).
// ---------------------------------------------------------------------------
__global__ __launch_bounds__(1024) void scan_kernel(const int* __restrict__ dur,
                                                    int* __restrict__ cum) {
    __shared__ int s[L_];
    const int b = blockIdx.x;
    const int i = threadIdx.x;
    int v = dur[(size_t)b * L_ + i];
    s[i] = v;
    __syncthreads();
    #pragma unroll
    for (int off = 1; off < L_; off <<= 1) {
        int add = (i >= off) ? s[i - off] : 0;
        __syncthreads();
        v += add;
        s[i] = v;
        __syncthreads();
    }
    cum[(size_t)b * L_ + i] = v;
}

// ---------------------------------------------------------------------------
// Length-regulate (searchsorted-right on cumsum) + bucketize + embedding add.
// Writes: bf16 regulated activations (predictor input) and final f32 output.
// One block per (b, t) row; 256 threads cover the 256 channels.
// ---------------------------------------------------------------------------
__global__ __launch_bounds__(256) void regulate_kernel(
    const float* __restrict__ x, const int* __restrict__ cum,
    const float* __restrict__ p_t, const float* __restrict__ e_t,
    const float* __restrict__ pbins, const float* __restrict__ ebins,
    const float* __restrict__ pemb, const float* __restrict__ eemb,
    unsigned short* __restrict__ regbf, float* __restrict__ out) {
    const int bt = blockIdx.x;
    const int b  = bt / T_;
    const int t  = bt - b * T_;
    __shared__ int sh[3];
    const int tid = threadIdx.x;
    if (tid == 0) {
        const int* c = cum + (size_t)b * L_;
        int lo = 0, hi = L_;
        while (lo < hi) { int m = (lo + hi) >> 1; if (c[m] <= t) lo = m + 1; else hi = m; }
        int idx = (lo < L_) ? lo : (L_ - 1);
        sh[0] = (t < c[L_ - 1]) ? idx : (idx | 0x80000000);
    } else if (tid == 1) {
        float v = p_t[bt];
        int lo = 0, hi = NBINS_;
        while (lo < hi) { int m = (lo + hi) >> 1; if (pbins[m] < v) lo = m + 1; else hi = m; }
        sh[1] = lo;
    } else if (tid == 2) {
        float v = e_t[bt];
        int lo = 0, hi = NBINS_;
        while (lo < hi) { int m = (lo + hi) >> 1; if (ebins[m] < v) lo = m + 1; else hi = m; }
        sh[2] = lo;
    }
    __syncthreads();
    const int  idx  = sh[0] & 0x7fffffff;
    const bool mask = (sh[0] >= 0);
    const int  d    = tid;
    float val = mask ? x[((size_t)b * L_ + idx) * D_ + d] : 0.0f;
    regbf[(size_t)bt * D_ + d] = f2bf(val);
    out[(size_t)bt * D_ + d]   = val + pemb[(size_t)sh[1] * D_ + d]
                                     + eemb[(size_t)sh[2] * D_ + d];
}

// ---------------------------------------------------------------------------
// Fused Conv1d(k=3, pad=1) + bias + LayerNorm + ReLU [+ Linear(256->1)+ReLU].
// 32 output rows x 256 output channels per block. bf16 WMMA, f32 accumulate.
// ---------------------------------------------------------------------------
template <bool PRED>
__global__ __launch_bounds__(256) void conv_ln_kernel(
    const unsigned short* __restrict__ in_bf,   // [rows, 256] bf16
    const unsigned short* __restrict__ wT,      // [3][256(out)][256(in)] bf16
    const float* __restrict__ bias,             // [256]
    const float* __restrict__ gamma,            // [256]
    const float* __restrict__ beta,             // [256]
    int Len,
    unsigned short* __restrict__ act_out,       // [rows, 256] bf16 (if !PRED)
    const float* __restrict__ lw,               // [256]  (if PRED)
    const float* __restrict__ lb,               // [1]    (if PRED)
    float* __restrict__ pred_out)               // [rows] (if PRED)
{
    __shared__ __align__(32) unsigned short sA[34 * 256];  // halo'd input tile
    __shared__ float sC[32 * 256];                          // f32 accum tile

    const int tid  = threadIdx.x;
    const int lane = tid & 31;
    const int wave = tid >> 5;
    const int row0  = blockIdx.x * 32;
    const int batch = row0 / Len;
    const int seq0  = row0 - batch * Len;

    const int rt    = wave & 1;      // row tile: 0/1
    const int cg    = wave >> 1;     // col group: 0..3 (64 cols each)
    const int lhalf = lane >> 4;     // 0: K 0..15, 1: K 16..31
    const int l15   = lane & 15;
    const int colb  = cg * 64 + l15;

    // Warm the weight stream for this wave (lowers to global_prefetch_b8).
    __builtin_prefetch(wT + (size_t)colb * D_, 0, 3);
    __builtin_prefetch(wT + (size_t)(D_ * D_ + colb * D_), 0, 3);
    __builtin_prefetch(wT + (size_t)(2 * D_ * D_ + colb * D_), 0, 3);

    // Stage 34 rows (with zero halo at sequence boundaries) x 256 ch of bf16.
    bool used_async = false;
    for (int ci = tid; ci < 34 * 32; ci += 256) {
        const int s   = ci >> 5;
        const int ch  = (ci & 31) << 3;
        const int seq = seq0 + s - 1;
        if (seq >= 0 && seq < Len) {
            const unsigned short* src = in_bf + ((size_t)(batch * Len + seq) * D_ + ch);
#if HAVE_ASYNC_LDS
            __builtin_amdgcn_global_load_async_to_lds_b128(
                (gas_v4i)(unsigned short*)src,
                (las_v4i)(&sA[s * 256 + ch]),
                0, 0);
            used_async = true;
#else
            *(uint4*)(&sA[s * 256 + ch]) = *(const uint4*)src;
#endif
        } else {
            uint4 z = {0u, 0u, 0u, 0u};
            *(uint4*)(&sA[s * 256 + ch]) = z;
        }
    }
    if (used_async) wait_asynccnt0();
    __syncthreads();

    v8f acc0 = {}, acc1 = {}, acc2 = {}, acc3 = {};

    for (int k = 0; k < 3; ++k) {
        const unsigned short* wk = wT + (size_t)k * D_ * D_ + (lhalf << 4);
        const unsigned short* ar = &sA[(rt * 16 + l15 + k) * 256 + (lhalf << 4)];
        #pragma unroll
        for (int cb = 0; cb < 8; ++cb) {
            const int c = cb << 5;
            v16bf af = *(const v16bf*)(ar + c);
            v16bf b0 = *(const v16bf*)(wk + (size_t)(colb +  0) * D_ + c);
            v16bf b1 = *(const v16bf*)(wk + (size_t)(colb + 16) * D_ + c);
            v16bf b2 = *(const v16bf*)(wk + (size_t)(colb + 32) * D_ + c);
            v16bf b3 = *(const v16bf*)(wk + (size_t)(colb + 48) * D_ + c);
            acc0 = __builtin_amdgcn_wmma_f32_16x16x32_bf16(false, af, false, b0, (short)0, acc0, false, false);
            acc1 = __builtin_amdgcn_wmma_f32_16x16x32_bf16(false, af, false, b1, (short)0, acc1, false, false);
            acc2 = __builtin_amdgcn_wmma_f32_16x16x32_bf16(false, af, false, b2, (short)0, acc2, false, false);
            acc3 = __builtin_amdgcn_wmma_f32_16x16x32_bf16(false, af, false, b3, (short)0, acc3, false, false);
        }
    }

    // C layout: VGPR r -> M=r (lanes 0-15) / M=r+8 (lanes 16-31), N=lane%16.
    const int Mb = rt * 16 + (lhalf << 3);
    {
        const float b0v = bias[cg * 64 +  0 + l15];
        const float b1v = bias[cg * 64 + 16 + l15];
        const float b2v = bias[cg * 64 + 32 + l15];
        const float b3v = bias[cg * 64 + 48 + l15];
        #pragma unroll
        for (int r = 0; r < 8; ++r) {
            sC[(Mb + r) * 256 + cg * 64 +  0 + l15] = acc0[r] + b0v;
            sC[(Mb + r) * 256 + cg * 64 + 16 + l15] = acc1[r] + b1v;
            sC[(Mb + r) * 256 + cg * 64 + 32 + l15] = acc2[r] + b2v;
            sC[(Mb + r) * 256 + cg * 64 + 48 + l15] = acc3[r] + b3v;
        }
    }
    __syncthreads();

    // LayerNorm + ReLU: 8 lanes per row, 32 channels per lane.
    const int row = tid >> 3;      // 0..31
    const int sub = tid & 7;       // 0..7
    const float* hrow = &sC[row * 256 + sub * 32];
    float s = 0.0f, s2 = 0.0f;
    #pragma unroll
    for (int i = 0; i < 32; ++i) { float v = hrow[i]; s += v; s2 += v * v; }
    #pragma unroll
    for (int m = 1; m < 8; m <<= 1) {
        s  += __shfl_xor(s,  m, 8);
        s2 += __shfl_xor(s2, m, 8);
    }
    const float mean = s * (1.0f / 256.0f);
    const float var  = s2 * (1.0f / 256.0f) - mean * mean;
    const float rstd = rsqrtf(var + 1e-5f);
    const int grow = row0 + row;
    const int ch0  = sub * 32;

    if (!PRED) {
        #pragma unroll
        for (int i = 0; i < 32; ++i) {
            const int ch = ch0 + i;
            float a = (hrow[i] - mean) * rstd * gamma[ch] + beta[ch];
            a = a > 0.0f ? a : 0.0f;
            act_out[(size_t)grow * D_ + ch] = f2bf(a);
        }
    } else {
        float p = 0.0f;
        #pragma unroll
        for (int i = 0; i < 32; ++i) {
            const int ch = ch0 + i;
            float a = (hrow[i] - mean) * rstd * gamma[ch] + beta[ch];
            a = a > 0.0f ? a : 0.0f;
            p += a * lw[ch];
        }
        #pragma unroll
        for (int m = 1; m < 8; m <<= 1) p += __shfl_xor(p, m, 8);
        if (sub == 0) {
            float pr = p + lb[0];
            pred_out[grow] = pr > 0.0f ? pr : 0.0f;
        }
    }
}

// ---------------------------------------------------------------------------
// Host launch sequence.
// ---------------------------------------------------------------------------
extern "C" void kernel_launch(void* const* d_in, const int* in_sizes, int n_in,
                              void* d_out, int out_size, void* d_ws, size_t ws_size,
                              hipStream_t stream) {
    (void)in_sizes; (void)n_in; (void)out_size; (void)ws_size;

    const float* x          = (const float*)d_in[0];
    const int*   durations  = (const int*)d_in[1];
    const float* p_target   = (const float*)d_in[2];
    const float* e_target   = (const float*)d_in[3];
    const float* pitch_bins = (const float*)d_in[5];
    const float* energy_bins= (const float*)d_in[6];
    const float* pitch_emb  = (const float*)d_in[7];
    const float* energy_emb = (const float*)d_in[8];

    // per-prefix param bases: dur=9, pit=19, en=29
    // +0 c1w, +1 c1b, +2 g1, +3 b1, +4 c2w, +5 c2b, +6 g2, +7 b2, +8 lw, +9 lb
    const int PB[3] = {9, 19, 29};

    // workspace layout (bytes, 512-aligned)
    char* ws = (char*)d_ws;
    int*            cum   = (int*)(ws + 0);                              // 128 KB
    unsigned short* xbf   = (unsigned short*)(ws + 131072);              // 16 MB
    unsigned short* regbf = (unsigned short*)(ws + 16908288);            // 48 MB
    unsigned short* h1bf  = (unsigned short*)(ws + 67239936);            // 48 MB
    unsigned short* wTbase= (unsigned short*)(ws + 117571584);           // 6 x 384 KB
    const size_t WSTRIDE = (size_t)3 * D_ * D_;                          // elems per weight

    float* out    = (float*)d_out;
    float* out_y  = out;                                   // [B,T,D]
    float* dur_p  = out + (size_t)B_ * T_ * D_;            // [B,L]
    float* pit_p  = dur_p + (size_t)B_ * L_;               // [B,T]
    float* en_p   = pit_p + (size_t)B_ * T_;               // [B,T]

    // 1) casts + weight transposes
    const int NX = B_ * L_ * D_;
    cvt_bf16_kernel<<<(NX + 255) / 256, 256, 0, stream>>>(x, xbf, NX);
    const int NW = 3 * D_ * D_;
    for (int p = 0; p < 3; ++p) {
        cvt_wT_kernel<<<(NW + 255) / 256, 256, 0, stream>>>(
            (const float*)d_in[PB[p] + 0], wTbase + (size_t)(2 * p + 0) * WSTRIDE);
        cvt_wT_kernel<<<(NW + 255) / 256, 256, 0, stream>>>(
            (const float*)d_in[PB[p] + 4], wTbase + (size_t)(2 * p + 1) * WSTRIDE);
    }

    // 2) duration cumsum + length regulate + bucketize/embed (final f32 output)
    scan_kernel<<<B_, 1024, 0, stream>>>(durations, cum);
    regulate_kernel<<<B_ * T_, 256, 0, stream>>>(
        x, cum, p_target, e_target, pitch_bins, energy_bins,
        pitch_emb, energy_emb, regbf, out_y);

    // 3) predictors: two fused conv passes each
    struct { const unsigned short* in; int len; int blocks; float* pred; int pb; int widx; }
    jobs[3] = {
        { xbf,   L_, (B_ * L_) / 32, dur_p, PB[0], 0 },
        { regbf, T_, (B_ * T_) / 32, pit_p, PB[1], 2 },
        { regbf, T_, (B_ * T_) / 32, en_p,  PB[2], 4 },
    };
    for (int j = 0; j < 3; ++j) {
        const int pb = jobs[j].pb;
        conv_ln_kernel<false><<<jobs[j].blocks, 256, 0, stream>>>(
            jobs[j].in, wTbase + (size_t)(jobs[j].widx + 0) * WSTRIDE,
            (const float*)d_in[pb + 1], (const float*)d_in[pb + 2], (const float*)d_in[pb + 3],
            jobs[j].len, h1bf, nullptr, nullptr, nullptr);
        conv_ln_kernel<true><<<jobs[j].blocks, 256, 0, stream>>>(
            h1bf, wTbase + (size_t)(jobs[j].widx + 1) * WSTRIDE,
            (const float*)d_in[pb + 5], (const float*)d_in[pb + 6], (const float*)d_in[pb + 7],
            jobs[j].len, nullptr, (const float*)d_in[pb + 8], (const float*)d_in[pb + 9],
            jobs[j].pred);
    }
}
